// SAN_85263690760552
// MI455X (gfx1250) — matrix operand order repaired and checked by reading, not compiled, not run on hardware
//
#include <hip/hip_runtime.h>
#include <math.h>

// ---------------------------------------------------------------------------
// CDNA5 WMMA GEMM:  C[M,N] = A[M,K] @ B[K,N] (+bias[N]), row-major fp32 I/O,
// f16 multiply / f32 accumulate via v_wmma_f32_16x16x32_f16.
// Block = 256 threads (8 wave32). Block tile 128x32; each wave computes a
// 16x32 strip as two 16x16 WMMA tiles sharing one A fragment.
// Requirements met by every GEMM in this pipeline:
//   K multiple of 32 (64/128/256/512); N, lda, ldb multiples of 4.
// Out-of-range rows/cols are handled by CLAMPING the staging addresses
// (branch-free, keeps all global_load_b128 in flight simultaneously);
// the resulting junk rows/cols of D are masked at the C store.
// ---------------------------------------------------------------------------
typedef __attribute__((ext_vector_type(16))) _Float16 v16h;
typedef __attribute__((ext_vector_type(8)))  _Float16 v8h;
typedef __attribute__((ext_vector_type(4)))  _Float16 v4h;
typedef __attribute__((ext_vector_type(8)))  float    v8f;

#define GEMM_BM 128
#define GEMM_BN 32
#define GEMM_KC 32
#define A_STRIDE (GEMM_KC + 8)   // 40 halves = 80 B: multiple of 16 B
#define B_STRIDE (GEMM_KC + 8)

__global__ __launch_bounds__(256)
void gemm_f16_wmma(const float* __restrict__ A, const float* __restrict__ B,
                   const float* __restrict__ bias, float* __restrict__ C,
                   int M, int N, int K, int lda, int ldb, int ldc)
{
    __shared__ __align__(16) _Float16 As[GEMM_BM * A_STRIDE];   // [m][k]
    __shared__ __align__(16) _Float16 BsT[GEMM_BN * B_STRIDE];  // [n][k] (transposed)
    const int tid  = threadIdx.x;
    const int lane = tid & 31;
    const int wv   = tid >> 5;          // 0..7 -> 16-row strip
    const int m0 = blockIdx.y * GEMM_BM;
    const int n0 = blockIdx.x * GEMM_BN;

    // clamped staging coordinates (branch-free loads)
    int rowA[4];
    #pragma unroll
    for (int t = 0; t < 4; ++t) {
        int r = (tid + t * 256) >> 3;
        int gm = m0 + r;
        rowA[t] = (gm < M) ? gm : (M - 1);
    }
    const int cA  = (tid & 7) << 2;            // A col within tile: 0,4,...,28
    const int rB  = tid >> 3;                  // B k-row 0..31
    const int cB  = (tid & 7) << 2;            // B n-col within tile
    int gnB = n0 + cB;
    gnB = (gnB <= N - 4) ? gnB : (N - 4);      // N multiple of 4 -> exact clamp

    v8f acc0 = {}, acc1 = {};
    for (int k0 = 0; k0 < K; k0 += GEMM_KC) {
        // ---- issue all 5 global_load_b128 unconditionally ----
        float4 va0 = *(const float4*)(A + (size_t)rowA[0] * lda + (k0 + cA));
        float4 va1 = *(const float4*)(A + (size_t)rowA[1] * lda + (k0 + cA));
        float4 va2 = *(const float4*)(A + (size_t)rowA[2] * lda + (k0 + cA));
        float4 va3 = *(const float4*)(A + (size_t)rowA[3] * lda + (k0 + cA));
        float4 vb  = *(const float4*)(B + (size_t)(k0 + rB) * ldb + gnB);

        // ---- convert + store to LDS ----
        const float4 va[4] = {va0, va1, va2, va3};
        #pragma unroll
        for (int t = 0; t < 4; ++t) {
            const int r = (tid + t * 256) >> 3;
            v4h h;
            h[0] = (_Float16)va[t].x; h[1] = (_Float16)va[t].y;
            h[2] = (_Float16)va[t].z; h[3] = (_Float16)va[t].w;
            *(v4h*)&As[r * A_STRIDE + cA] = h;
        }
        BsT[(cB + 0) * B_STRIDE + rB] = (_Float16)vb.x;
        BsT[(cB + 1) * B_STRIDE + rB] = (_Float16)vb.y;
        BsT[(cB + 2) * B_STRIDE + rB] = (_Float16)vb.z;
        BsT[(cB + 3) * B_STRIDE + rB] = (_Float16)vb.w;
        __syncthreads();

        // A fragment: lanes 0-15 K=0..15; lanes 16-31 K=16..31 (contiguous, matches B)
        const int mloc  = wv * 16 + (lane & 15);
        const int koffA = (lane >= 16) ? 16 : 0;
        const v8h a0 = *(const v8h*)&As[mloc * A_STRIDE + koffA];
        const v8h a1 = *(const v8h*)&As[mloc * A_STRIDE + koffA + 8];
        v16h af;
        #pragma unroll
        for (int j = 0; j < 8; ++j) { af[j] = a0[j]; af[j + 8] = a1[j]; }

        // B fragments: lanes 0-15 K=0..15, lanes 16-31 K=16..31; contiguous in BsT
        const int koffB = (lane >= 16) ? 16 : 0;
        const int nl    = lane & 15;
        const v8h b0lo = *(const v8h*)&BsT[nl * B_STRIDE + koffB];
        const v8h b0hi = *(const v8h*)&BsT[nl * B_STRIDE + koffB + 8];
        const v8h b1lo = *(const v8h*)&BsT[(16 + nl) * B_STRIDE + koffB];
        const v8h b1hi = *(const v8h*)&BsT[(16 + nl) * B_STRIDE + koffB + 8];
        v16h bf0, bf1;
        #pragma unroll
        for (int j = 0; j < 8; ++j) {
            bf0[j] = b0lo[j]; bf0[j + 8] = b0hi[j];
            bf1[j] = b1lo[j]; bf1[j + 8] = b1hi[j];
        }

        acc0 = __builtin_amdgcn_wmma_f32_16x16x32_f16(false, af, false, bf0,
                                                      (short)0, acc0, false, false);
        acc1 = __builtin_amdgcn_wmma_f32_16x16x32_f16(false, af, false, bf1,
                                                      (short)0, acc1, false, false);
        __syncthreads();
    }

    // C/D layout: lane 0-15 -> N=lane, M rows 0..7; lane 16-31 -> N=lane-16, M 8..15
    const int mb  = m0 + wv * 16 + ((lane >> 4) << 3);
    const int n_a = n0 + (lane & 15);
    const int n_b = n_a + 16;
    const float bva = (bias && n_a < N) ? bias[n_a] : 0.f;
    const float bvb = (bias && n_b < N) ? bias[n_b] : 0.f;
    #pragma unroll
    for (int r = 0; r < 8; ++r) {
        const int m = mb + r;
        if (m < M) {
            if (n_a < N) C[(size_t)m * ldc + n_a] = acc0[r] + bva;
            if (n_b < N) C[(size_t)m * ldc + n_b] = acc1[r] + bvb;
        }
    }
}

// ---------------------------------------------------------------------------
// Small helper kernels
// ---------------------------------------------------------------------------
__global__ void fill_k(float* __restrict__ p, float v, long long n)
{
    long long i = (long long)blockIdx.x * blockDim.x + threadIdx.x;
    if (i < n) p[i] = v;
}

// el[n,h] = sum_d hs[n, h*D+d] * aw[h,d]      (hs: [Nn, H*D], aw: [H,D])
__global__ void attn_scores_k(const float* __restrict__ hs, const float* __restrict__ aw,
                              float* __restrict__ out, int Nn, int H, int D)
{
    int i = blockIdx.x * blockDim.x + threadIdx.x;
    if (i >= Nn * H) return;
    int n = i / H, h = i % H;
    const float* hp = hs + (size_t)n * H * D + h * D;
    const float* ap = aw + h * D;
    float acc = 0.f;
    for (int d = 0; d < D; ++d) acc += hp[d] * ap[d];
    out[i] = acc;
}

// V[k,h] = sum_d W[k, h*D+d] * aw[h,d]        (W: [Kd, H*D])  — attention fold
__global__ void fold_attn_k(const float* __restrict__ W, const float* __restrict__ aw,
                            float* __restrict__ V, int Kd, int H, int D)
{
    int i = blockIdx.x * blockDim.x + threadIdx.x;
    if (i >= Kd * H) return;
    int k = i / H, h = i % H;
    const float* wr = W + (size_t)k * H * D + h * D;
    const float* ap = aw + h * D;
    float acc = 0.f;
    for (int d = 0; d < D; ++d) acc += wr[d] * ap[d];
    V[k * H + h] = acc;
}

// out[n,h] = sum_k X[n,k] * V[k,h]   (tiny N=H=4 matmul: er = hpd @ fold(W,ar))
__global__ void mm_small_k(const float* __restrict__ X, const float* __restrict__ V,
                           float* __restrict__ out, int Nn, int Kd, int H)
{
    int i = blockIdx.x * blockDim.x + threadIdx.x;
    if (i >= Nn * H) return;
    int n = i / H, h = i % H;
    const float* xp = X + (size_t)n * Kd;
    float acc = 0.f;
    for (int k = 0; k < Kd; ++k) acc += xp[k] * V[k * H + h];
    out[i] = acc;
}

// y = elu(x + b[c % HD]);  x may alias y
__global__ void elu_bias_k(const float* __restrict__ x, const float* __restrict__ b,
                           float* __restrict__ y, long long n, int HD)
{
    long long i = (long long)blockIdx.x * blockDim.x + threadIdx.x;
    if (i >= n) return;
    float v = x[i] + b[(int)(i % HD)];
    y[i] = (v > 0.f) ? v : (expf(v) - 1.f);
}

// ---------------------------------------------------------------------------
// Edge kernels (H == 4 hard-wired).  All segment reductions via L2 atomics:
// every rst accumulator (<=102 MB) is resident in the 192 MB L2.
// ---------------------------------------------------------------------------
__device__ __forceinline__ float leaky02(float x) { return x > 0.f ? x : 0.2f * x; }

// monotone signed-int max / unsigned-int min trick -> float atomic max
__device__ __forceinline__ void atomicMaxF(float* addr, float v)
{
    if (v >= 0.f) atomicMax((int*)addr, __float_as_int(v));
    else          atomicMin((unsigned int*)addr, __float_as_uint(v));
}

__global__ void edge_max_k(const int* __restrict__ src, const int* __restrict__ dst,
                           const float* __restrict__ el, const float* __restrict__ er,
                           float* __restrict__ m, int E)
{
    int i = blockIdx.x * blockDim.x + threadIdx.x;
    if (i >= E * 4) return;
    int e = i >> 2, h = i & 3;
    int s = src[e], d = dst[e];
    float x = leaky02(el[s * 4 + h] + er[d * 4 + h]);
    atomicMaxF(&m[d * 4 + h], x);
}

__global__ void edge_sum_k(const int* __restrict__ src, const int* __restrict__ dst,
                           const float* __restrict__ el, const float* __restrict__ er,
                           const float* __restrict__ m, float* __restrict__ ssum, int E)
{
    int i = blockIdx.x * blockDim.x + threadIdx.x;
    if (i >= E * 4) return;
    int e = i >> 2, h = i & 3;
    int s = src[e], d = dst[e];
    float x = leaky02(el[s * 4 + h] + er[d * 4 + h]);
    atomicAdd(&ssum[d * 4 + h], expf(x - m[d * 4 + h]));
}

// rst[dst, c..c+3] += alpha(e, c/D) * hs[src, c..c+3]; attention recomputed
__global__ void edge_agg_k(const int* __restrict__ src, const int* __restrict__ dst,
                           const float* __restrict__ el, const float* __restrict__ er,
                           const float* __restrict__ m, const float* __restrict__ ssum,
                           const float* __restrict__ hs, float* __restrict__ rst,
                           int E, int HD, int D)
{
    int cpe = HD >> 2;                              // float4 chunks per edge
    long long i = (long long)blockIdx.x * blockDim.x + threadIdx.x;
    if (i >= (long long)E * cpe) return;
    int e  = (int)(i / cpe);
    int c  = (int)(i % cpe) << 2;
    int h  = c / D;
    int s  = src[e], d = dst[e];
    float x = leaky02(el[s * 4 + h] + er[d * 4 + h]);
    float alpha = expf(x - m[d * 4 + h]) / ssum[d * 4 + h];
    const float4 hv = *(const float4*)(hs + (size_t)s * HD + c);
    float* rp = rst + (size_t)d * HD + c;
    atomicAdd(rp + 0, alpha * hv.x);
    atomicAdd(rp + 1, alpha * hv.y);
    atomicAdd(rp + 2, alpha * hv.z);
    atomicAdd(rp + 3, alpha * hv.w);
}

// ---------------------------------------------------------------------------
// Host orchestration
// ---------------------------------------------------------------------------
static inline unsigned nblk(long long n) { return (unsigned)((n + 255) / 256); }

extern "C" void kernel_launch(void* const* d_in, const int* in_sizes, int n_in,
                              void* d_out, int out_size, void* d_ws, size_t ws_size,
                              hipStream_t stream)
{
    (void)n_in; (void)ws_size;
    // ---- inputs (setup_inputs order) ----
    const float* hp  = (const float*)d_in[0];
    const float* hq  = (const float*)d_in[1];
    const float* hk  = (const float*)d_in[2];
    const float* ha  = (const float*)d_in[3];
    const int *msrc = (const int*)d_in[4],  *mdst = (const int*)d_in[5];
    const int *qsrc = (const int*)d_in[6],  *qdst = (const int*)d_in[7];
    const int *ksrc = (const int*)d_in[8],  *kdst = (const int*)d_in[9];
    const int *asrc = (const int*)d_in[10], *adst = (const int*)d_in[11];
    const float *W0=(const float*)d_in[12], *al0=(const float*)d_in[13], *ar0=(const float*)d_in[14], *b0=(const float*)d_in[15];
    const float *W1=(const float*)d_in[16], *al1=(const float*)d_in[17], *ar1=(const float*)d_in[18], *b1=(const float*)d_in[19];
    const float *Wq =(const float*)d_in[20], *alq=(const float*)d_in[21], *arq=(const float*)d_in[22], *bq =(const float*)d_in[23];
    const float *Wkw=(const float*)d_in[24], *alk=(const float*)d_in[25], *ark=(const float*)d_in[26], *bkw=(const float*)d_in[27];
    const float *Wat=(const float*)d_in[28], *ala=(const float*)d_in[29], *ara=(const float*)d_in[30], *bat=(const float*)d_in[31];
    const float *Wpnn_a=(const float*)d_in[32], *Wpn_a=(const float*)d_in[33];
    const float *Wpnn  =(const float*)d_in[34], *Wpn  =(const float*)d_in[35];
    const float *Wpred =(const float*)d_in[36], *bpred=(const float*)d_in[37];
    float* out = (float*)d_out;

    const int IN = 128, ATT = 64;
    const int NP  = in_sizes[0] / IN;
    const int NQ  = in_sizes[1] / ATT;
    const int NKN = in_sizes[2] / ATT;
    const int NA  = in_sizes[3] / ATT;
    const int EM = in_sizes[4], EQ = in_sizes[6], EK = in_sizes[8], EA = in_sizes[10];
    const int OUT = in_sizes[37];
    (void)out_size;

    // ---- workspace carve-up (floats) ----
    float* wsf = (float*)d_ws;
    size_t off = 0;
    float* R_rst  = wsf + off; off += (size_t)NP * 512;     // segment-sum accumulator
    float* R_hs   = wsf + off; off += (size_t)30000 * 512;  // hs_src (meta hs NP*128 fits)
    float* R_h    = wsf + off; off += (size_t)NP * 128;     // node features between layers
    float* R_hpd  = wsf + off; off += (size_t)NP * 64;      // proj_pnn_a(hp)
    float* R_x    = wsf + off; off += (size_t)30000 * 64;   // proj_pn_a(src feats)
    float* R_hcat = wsf + off; off += (size_t)NP * 256;     // concat buffer
    float* R_el   = wsf + off; off += (size_t)NP * 4;
    float* R_er   = wsf + off; off += (size_t)NP * 4;
    float* R_m    = wsf + off; off += (size_t)NP * 4;
    float* R_s    = wsf + off; off += (size_t)NP * 4;
    float* R_V    = wsf + off; off += 512;                  // folded attention vec

    auto gemm = [&](const float* A, const float* B, const float* bias, float* C,
                    int M, int N, int K, int lda, int ldb, int ldc) {
        dim3 g((N + GEMM_BN - 1) / GEMM_BN, (M + GEMM_BM - 1) / GEMM_BM);
        gemm_f16_wmma<<<g, 256, 0, stream>>>(A, B, bias, C, M, N, K, lda, ldb, ldc);
    };

    auto run_edges = [&](const int* src, const int* dst, int E, int HD, int D) {
        fill_k<<<nblk((long long)NP * 4), 256, 0, stream>>>(R_m, -INFINITY, (long long)NP * 4);
        fill_k<<<nblk((long long)NP * 4), 256, 0, stream>>>(R_s, 0.f, (long long)NP * 4);
        fill_k<<<nblk((long long)NP * HD), 256, 0, stream>>>(R_rst, 0.f, (long long)NP * HD);
        edge_max_k<<<nblk((long long)E * 4), 256, 0, stream>>>(src, dst, R_el, R_er, R_m, E);
        edge_sum_k<<<nblk((long long)E * 4), 256, 0, stream>>>(src, dst, R_el, R_er, R_m, R_s, E);
        edge_agg_k<<<nblk((long long)E * (HD >> 2)), 256, 0, stream>>>(
            src, dst, R_el, R_er, R_m, R_s, R_hs, R_rst, E, HD, D);
    };

    // self-graph GAT layer: feat[NP,128] -> hout[NP,128]
    auto meta_layer = [&](const float* feat, const float* W, const float* al,
                          const float* ar, const float* b, float* hout) {
        gemm(feat, W, nullptr, R_hs, NP, 128, 128, 128, 128, 128);        // hs = feat@W
        attn_scores_k<<<nblk((long long)NP * 4), 256, 0, stream>>>(R_hs, al, R_el, NP, 4, 32);
        attn_scores_k<<<nblk((long long)NP * 4), 256, 0, stream>>>(R_hs, ar, R_er, NP, 4, 32);
        run_edges(msrc, mdst, EM, 128, 32);
        elu_bias_k<<<nblk((long long)NP * 128), 256, 0, stream>>>(R_rst, b, hout,
                                                                  (long long)NP * 128, 128);
    };

    // bipartite GAT: x[Ns,64] src feats, hpd dst feats; writes hcat[:,co:co+64]
    auto bip_layer = [&](const float* x, int Ns, const int* src, const int* dst, int E,
                         const float* Wg, const float* alg, const float* arg,
                         const float* bg, int co) {
        gemm(x, Wpn_a, nullptr, R_x, Ns, 64, 64, 64, 64, 64);             // proj_pn_a
        gemm(R_x, Wg, nullptr, R_hs, Ns, 512, 64, 64, 512, 512);          // hs = x'@Wg
        attn_scores_k<<<nblk((long long)Ns * 4), 256, 0, stream>>>(R_hs, alg, R_el, Ns, 4, 128);
        // er = (hpd@Wg * ar).sum  ==  hpd @ fold(Wg, ar)   (avoids materializing hd)
        fold_attn_k<<<nblk(64 * 4), 256, 0, stream>>>(Wg, arg, R_V, 64, 4, 128);
        mm_small_k<<<nblk((long long)NP * 4), 256, 0, stream>>>(R_hpd, R_V, R_er, NP, 64, 4);
        run_edges(src, dst, E, 512, 128);
        elu_bias_k<<<nblk((long long)NP * 512), 256, 0, stream>>>(R_rst, bg, R_rst,
                                                                  (long long)NP * 512, 512);
        gemm(R_rst, Wpn, nullptr, R_hcat + co, NP, 64, 512, 512, 64, 256); // project into concat
    };

    // ---- pipeline ----
    meta_layer(hp,  W0, al0, ar0, b0, R_h);                               // layer 1
    meta_layer(R_h, W1, al1, ar1, b1, R_h);                               // layer 2

    gemm(hp,  Wpnn_a, nullptr, R_hpd, NP, 64, 128, 128, 64, 64);          // dst feats
    gemm(R_h, Wpnn,   nullptr, R_hcat + 0, NP, 64, 128, 128, 64, 256);    // hcat[:,0:64]

    bip_layer(hq, NQ,  qsrc, qdst, EQ, Wq,  alq, arq, bq,  64);
    bip_layer(hk, NKN, ksrc, kdst, EK, Wkw, alk, ark, bkw, 128);
    bip_layer(ha, NA,  asrc, adst, EA, Wat, ala, ara, bat, 192);

    gemm(R_hcat, Wpred, bpred, out, NP, OUT, 256, 256, OUT, OUT);         // final logits
}
